// DynamicGraphBuilder_15307263443518
// MI455X (gfx1250) — matrix operand order repaired and checked by reading, not compiled, not run on hardware
//
#include <hip/hip_runtime.h>
#include <hip/hip_bf16.h>

typedef __attribute__((ext_vector_type(16))) _Float16 v16h;
typedef __attribute__((ext_vector_type(8)))  _Float16 v8h;
typedef __attribute__((ext_vector_type(8)))  float    v8f;

#define MAX_NODES 1024
#define FEAT 64
#define HID 32
#define BB 32
#define SS 4096
// edges E = BB*SS = 131072; tiles = E/16 = 8192

// ---------------- Kernel 1: zero entire output ----------------
__global__ void dgb_zero_f4(float4* __restrict__ p, long n4) {
    long i = (long)blockIdx.x * blockDim.x + threadIdx.x;
    long stride = (long)gridDim.x * blockDim.x;
    float4 z = {0.f, 0.f, 0.f, 0.f};
    for (; i < n4; i += stride) p[i] = z;
}

// ---------------- Kernel 2: prep f16 embeddings + swizzled W1 B-fragments ----
// bfrag layout: frag f = t*4+c (t = N-tile 0..1, c = K-chunk 0..3)
//   half q = ((f*32 + lane)*16 + i)  <-  W1[(32c + (i&7) + (i>=8?16:0) + (lane>=16?8:0))*32 + 16t + (lane&15)]
__global__ void dgb_prep(const float* __restrict__ node_emb,
                         const float* __restrict__ W1,
                         _Float16* __restrict__ emb_h,
                         _Float16* __restrict__ bfrag) {
    int tid = blockIdx.x * blockDim.x + threadIdx.x;
    if (tid < MAX_NODES * FEAT) {
        emb_h[tid] = (_Float16)node_emb[tid];
    } else {
        int q = tid - MAX_NODES * FEAT;
        if (q < 8 * 32 * 16) {
            int f = q >> 9;            // 0..7
            int l = (q >> 4) & 31;     // lane
            int i = q & 15;            // half index
            int t = f >> 2, c = f & 3;
            int krow = 32 * c + (i & 7) + ((i >= 8) ? 16 : 0) + ((l >= 16) ? 8 : 0);
            int n = 16 * t + (l & 15);
            bfrag[q] = (_Float16)W1[krow * HID + n];
        }
    }
}

// ---------------- Kernel 3: scatter-add flow features into node_features ----
__global__ void dgb_scatter_add(const float* __restrict__ ff,
                                const int* __restrict__ dst_ips,
                                float* __restrict__ nf) {
    int tid = blockIdx.x * blockDim.x + threadIdx.x;   // < B*S*F = 8388608
    int f  = tid & (FEAT - 1);
    int bs = tid >> 6;              // b*S + s
    int b  = bs >> 12;              // /S
    int d  = dst_ips[bs];
    unsafeAtomicAdd(&nf[((b << 10) + d) * FEAT + f], ff[tid]);
}

// ---------------- Kernel 4: L2-normalize node rows (one wave per node) ------
__global__ void dgb_normalize(float* __restrict__ nf) {
    int lane = threadIdx.x & 31;
    int node = (blockIdx.x * blockDim.x + threadIdx.x) >> 5;  // < B*N = 32768
    float2* p = (float2*)(nf + node * FEAT) + lane;
    float2 v = *p;
    float ss = v.x * v.x + v.y * v.y;
    ss += __shfl_xor(ss, 1, 32);
    ss += __shfl_xor(ss, 2, 32);
    ss += __shfl_xor(ss, 4, 32);
    ss += __shfl_xor(ss, 8, 32);
    ss += __shfl_xor(ss, 16, 32);
    float inv = 1.0f / fmaxf(sqrtf(ss), 1e-12f);
    v.x *= inv; v.y *= inv;
    *p = v;
}

__device__ __forceinline__ float dgb_sigmoid(float x) {
    return 1.0f / (1.0f + __expf(-x));
}

// ---------------- Kernel 5: edge MLP via WMMA + adjacency scatter -----------
// One wave handles a tile of 16 edges. X = [emb[src] | emb[dst]] (16x128 f16),
// h = relu(X@W1 + b1) via 8x v_wmma_f32_16x16x32_f16, then h@W2, sigmoid.
__global__ void dgb_edge_wmma(const _Float16* __restrict__ emb_h,
                              const _Float16* __restrict__ bfrag,
                              const int* __restrict__ src_ips,
                              const int* __restrict__ dst_ips,
                              const float* __restrict__ vols,
                              const float* __restrict__ b1,
                              const float* __restrict__ W2,
                              const float* __restrict__ b2,
                              float* __restrict__ adj) {
    const int lane = threadIdx.x & 31;
    const int wave = threadIdx.x >> 5;
    const int tile = blockIdx.x * 8 + wave;     // 8192 tiles
    const int e0 = tile * 16;
    const int m = lane & 15;                    // A row handled by this lane
    const int b = e0 >> 12;                     // edge block's batch (S=4096 | 16)
    const int src = src_ips[e0 + m];
    const int dst = dst_ips[e0 + m];
    const int kbase = (lane < 16) ? 0 : 8;      // 16-bit A-matrix lane split

    // B fragments: 8 contiguous v16h loads per lane (pre-swizzled)
    v16h bf[8];
#pragma unroll
    for (int f = 0; f < 8; ++f)
        bf[f] = *(const v16h*)(bfrag + (f * 32 + lane) * 16);

    v8f acc0 = {};  // h columns 0..15
    v8f acc1 = {};  // h columns 16..31
#pragma unroll
    for (int c = 0; c < 4; ++c) {
        const int node = (c < 2) ? src : dst;
        const int ofs  = (c & 1) * 32;
        const _Float16* row = emb_h + node * FEAT + ofs + kbase;
        v8h lo = *(const v8h*)(row);        // K_local = kbase..kbase+7
        v8h hi = *(const v8h*)(row + 16);   // K_local = 16+kbase..23+kbase
        v16h a;
#pragma unroll
        for (int i = 0; i < 8; ++i) { a[i] = lo[i]; a[i + 8] = hi[i]; }
        acc0 = __builtin_amdgcn_wmma_f32_16x16x32_f16(false, a, false, bf[c],
                                                      (short)0, acc0, false, false);
        acc1 = __builtin_amdgcn_wmma_f32_16x16x32_f16(false, a, false, bf[4 + c],
                                                      (short)0, acc1, false, false);
    }

    // Bias + ReLU + h@W2.  C/D layout: lane l slot j -> h[j + (l>=16?8:0)][l&15 (+16t)]
    const float b1lo = b1[m],  b1hi = b1[m + 16];
    const float w2lo = W2[m],  w2hi = W2[m + 16];
    float dm = 0.0f;
#pragma unroll
    for (int j = 0; j < 8; ++j) {
        float h0 = fmaxf(acc0[j] + b1lo, 0.0f);
        float h1 = fmaxf(acc1[j] + b1hi, 0.0f);
        float v = h0 * w2lo + h1 * w2hi;
        // reduce over each 16-lane half-wave (xor masks stay within the half)
        v += __shfl_xor(v, 1, 32);
        v += __shfl_xor(v, 2, 32);
        v += __shfl_xor(v, 4, 32);
        v += __shfl_xor(v, 8, 32);
        dm = ((lane & 7) == j) ? v : dm;
    }

    // Writer lanes: 0..7 -> edges 0..7, 16..23 -> edges 8..15
    if ((lane & 15) < 8) {
        const int me = (lane & 7) + ((lane >= 16) ? 8 : 0);
        const int e = e0 + me;
        const float edge_w = dgb_sigmoid(dm + b2[0]);
        const float vol_w  = dgb_sigmoid(vols[e] * 0.001f);
        const float w = edge_w * vol_w;
        const int si = src_ips[e];
        const int di = dst_ips[e];
        float* A = adj + (size_t)b * MAX_NODES * MAX_NODES;
        A[si * MAX_NODES + di] = w;
        A[di * MAX_NODES + si] = w;
    }
}

extern "C" void kernel_launch(void* const* d_in, const int* in_sizes, int n_in,
                              void* d_out, int out_size, void* d_ws, size_t ws_size,
                              hipStream_t stream) {
    const float* flow_features = (const float*)d_in[0];
    const int*   src_ips       = (const int*)d_in[1];
    const int*   dst_ips       = (const int*)d_in[2];
    const float* flow_volumes  = (const float*)d_in[3];
    const float* node_emb      = (const float*)d_in[4];
    const float* W1            = (const float*)d_in[5];
    const float* b1            = (const float*)d_in[6];
    const float* W2            = (const float*)d_in[7];
    const float* b2            = (const float*)d_in[8];

    float* out = (float*)d_out;
    float* nf  = out;                              // (32,1024,64)
    float* adj = out + BB * MAX_NODES * FEAT;      // (32,1024,1024)

    _Float16* emb_h = (_Float16*)d_ws;             // 65536 halves
    _Float16* bfrag = emb_h + MAX_NODES * FEAT;    // 4096 halves

    // 1) zero outputs (out_size = 35,651,584 floats, divisible by 4)
    dgb_zero_f4<<<2048, 256, 0, stream>>>((float4*)out, (long)out_size / 4);

    // 2) prep f16 embeddings + W1 WMMA B-fragments (65536 + 4096 = 272*256)
    dgb_prep<<<272, 256, 0, stream>>>(node_emb, W1, emb_h, bfrag);

    // 3) scatter-add flow features (B*S*F = 8,388,608 threads)
    dgb_scatter_add<<<32768, 256, 0, stream>>>(flow_features, dst_ips, nf);

    // 4) normalize node rows (B*N waves = 32768 -> 4096 blocks)
    dgb_normalize<<<4096, 256, 0, stream>>>(nf);

    // 5) edge MLP + adjacency scatter (8192 tiles, 8 waves/block)
    dgb_edge_wmma<<<1024, 256, 0, stream>>>(emb_h, bfrag, src_ips, dst_ips,
                                            flow_volumes, b1, W2, b2, adj);
}
